// LightningDecoderLayer_25640954757725
// MI455X (gfx1250) — compile-verified, matrix operand order checked by reading, not run on hardware
//
#include <hip/hip_runtime.h>
#include <hip/hip_bf16.h>
#include <cstdint>

// ---------------- problem constants ----------------
#define BB     2
#define TT     2048
#define HID    1024
#define HH     16
#define DD     64
#define KD     1024
#define MM     (BB*TT)        // 4096 rows
#define NBIG   4096           // q|k|v|g packed columns
#define CONVW  4
#define EPSR   1e-6f

// ---------------- WMMA types ----------------
typedef __attribute__((ext_vector_type(16))) __bf16 v16bf;
typedef __attribute__((ext_vector_type(8)))  float  v8f;
typedef __attribute__((ext_vector_type(4)))  int    v4i;

// Async global->LDS path (CDNA5): builtin confirmed present; signature is
// (v4i addrspace(1)* gaddr, v4i addrspace(3)* laddr, imm offset, imm cpol).
#if __has_builtin(__builtin_amdgcn_global_load_async_to_lds_b128) && \
    __has_builtin(__builtin_amdgcn_s_wait_asynccnt)
#define USE_ASYNC_LDS 1
typedef __attribute__((address_space(1))) v4i gv4i_t;
typedef __attribute__((address_space(3))) v4i lv4i_t;
#else
#define USE_ASYNC_LDS 0
#endif

__device__ __forceinline__ unsigned short f2bf(float f) {
    union { float f; unsigned u; } x; x.f = f;
    unsigned u = x.u;
    unsigned r = u + 0x7FFFu + ((u >> 16) & 1u);   // round-to-nearest-even
    return (unsigned short)(r >> 16);
}
__device__ __forceinline__ float sigmoidf_(float x) { return 1.0f / (1.0f + __expf(-x)); }
__device__ __forceinline__ float siluf_(float x)    { return x * sigmoidf_(x); }

// ---------------- f32 -> bf16 convert ----------------
__global__ void cvt_f32_bf16(const float* __restrict__ src, unsigned short* __restrict__ dst, int n) {
    int i = blockIdx.x * 256 + threadIdx.x;
    if (i < n) dst[i] = f2bf(src[i]);
}

// ---------------- WMMA bf16 GEMM: C[M,N] = A[M,K] * Bw[N,K]^T ----------------
// tile: 128(M) x 128(N) per 256-thread block (8 waves).
// wave (wm = wave&3, wn = wave>>2) owns a 32x64 sub-tile: 2 A-frags x 4 B-frags
// -> 8 v_wmma_f32_16x16x32_bf16 per K-step of 32.
#define TILE_M 128
#define TILE_N 128
#define TILE_K 32
#define LDS_STRIDE 40   // ushorts per row (80B, keeps b128 stores 16B-aligned)

__global__ void __launch_bounds__(256)
wmma_gemm_bf16(const unsigned short* __restrict__ A,
               const unsigned short* __restrict__ Bw,
               float* __restrict__ C, int M, int N, int K)
{
    __shared__ unsigned short As[TILE_M * LDS_STRIDE];
    __shared__ unsigned short Bs[TILE_N * LDS_STRIDE];

    const int tid  = threadIdx.x;
    const int wave = tid >> 5;
    const int lane = tid & 31;
    const int half = lane >> 4;         // fragment K-half select (ISA 16-bit layout)
    const int r16  = lane & 15;

    const int wm = wave & 3;            // 4 M-strips of 32 rows
    const int wn = wave >> 2;           // 2 N-strips of 64 cols

    const int tileN = blockIdx.x * TILE_N;
    const int tileM = blockIdx.y * TILE_M;

    v8f acc[2][4];
    #pragma unroll
    for (int f = 0; f < 2; ++f)
        #pragma unroll
        for (int n = 0; n < 4; ++n) acc[f][n] = (v8f){0,0,0,0,0,0,0,0};

    // staging: both tiles are 128 rows x 32 cols bf16 -> 32B per thread each
    const int sRow = tid >> 1, sSeg = tid & 1;

    for (int k0 = 0; k0 < K; k0 += TILE_K) {
        const unsigned short* srcA = A  + (size_t)(tileM + sRow) * K + k0 + sSeg * 16;
        const unsigned short* srcB = Bw + (size_t)(tileN + sRow) * K + k0 + sSeg * 16;
        unsigned short* dstA = &As[sRow * LDS_STRIDE + sSeg * 16];
        unsigned short* dstB = &Bs[sRow * LDS_STRIDE + sSeg * 16];
#if USE_ASYNC_LDS
        __builtin_amdgcn_global_load_async_to_lds_b128((gv4i_t*)srcA,       (lv4i_t*)dstA,       0, 0);
        __builtin_amdgcn_global_load_async_to_lds_b128((gv4i_t*)(srcA + 8), (lv4i_t*)(dstA + 8), 0, 0);
        __builtin_amdgcn_global_load_async_to_lds_b128((gv4i_t*)srcB,       (lv4i_t*)dstB,       0, 0);
        __builtin_amdgcn_global_load_async_to_lds_b128((gv4i_t*)(srcB + 8), (lv4i_t*)(dstB + 8), 0, 0);
        __builtin_amdgcn_s_wait_asynccnt(0);
#else
        {
            uint4 a0 = *(const uint4*)(srcA);
            uint4 a1 = *(const uint4*)(srcA + 8);
            uint4 b0 = *(const uint4*)(srcB);
            uint4 b1 = *(const uint4*)(srcB + 8);
            *(uint4*)(dstA)     = a0;
            *(uint4*)(dstA + 8) = a1;
            *(uint4*)(dstB)     = b0;
            *(uint4*)(dstB + 8) = b1;
        }
#endif
        if (k0 + TILE_K < K) {
            __builtin_prefetch((const void*)(srcA + TILE_K), 0, 1);
            __builtin_prefetch((const void*)(srcB + TILE_K), 0, 1);
        }
        __syncthreads();

        const __bf16* Ab = (const __bf16*)As;
        const __bf16* Bb = (const __bf16*)Bs;

        // A fragments: 16x32 (rows wm*32 + f*16 .. +15); contiguous 8+8 bf16 per lane
        v16bf afrag[2];
        #pragma unroll
        for (int f = 0; f < 2; ++f) {
            const int arow = wm * 32 + f * 16 + r16;
            #pragma unroll
            for (int p = 0; p < 8; ++p) {
                const int kk = ((p < 4) ? 0 : 16) + half * 8 + (p & 3) * 2;
                afrag[f][2*p]   = Ab[arow * LDS_STRIDE + kk];
                afrag[f][2*p+1] = Ab[arow * LDS_STRIDE + kk + 1];
            }
        }
        // B fragments (32x16) + 8 WMMAs
        #pragma unroll
        for (int nt = 0; nt < 4; ++nt) {
            v16bf bfrag;
            const int brow = wn * 64 + nt * 16 + r16;
            #pragma unroll
            for (int p = 0; p < 8; ++p) {
                const int kk = ((p < 4) ? 0 : 16) + half * 8 + (p & 3) * 2;
                bfrag[2*p]   = Bb[brow * LDS_STRIDE + kk];
                bfrag[2*p+1] = Bb[brow * LDS_STRIDE + kk + 1];
            }
            #pragma unroll
            for (int f = 0; f < 2; ++f)
                acc[f][nt] = __builtin_amdgcn_wmma_f32_16x16x32_bf16(
                    false, afrag[f], false, bfrag, (short)0, acc[f][nt], false, false);
        }
        __syncthreads();
    }

    // store D: lane(0-15)=N col, lane(16-31)=N col with M+8; vgpr r = M row
    const int nl = lane & 15;
    const int mh = lane >> 4;
    #pragma unroll
    for (int f = 0; f < 2; ++f) {
        #pragma unroll
        for (int r = 0; r < 8; ++r) {
            const int m = tileM + wm * 32 + f * 16 + mh * 8 + r;
            float* crow = C + (size_t)m * N + tileN + wn * 64;
            #pragma unroll
            for (int nt = 0; nt < 4; ++nt)
                crow[nt * 16 + nl] = acc[f][nt][r];
        }
    }
}

// ---------------- depthwise conv(4)+SiLU+RoPE+L2norm for q/k ----------------
// grid = B*T*H blocks, block = 64 (d index)
__global__ void conv_rope_l2(const float* __restrict__ pre, int colbase,
                             const float* __restrict__ cw, const float* __restrict__ cb,
                             float* __restrict__ out)
{
    __shared__ float sv[DD];
    __shared__ float red[DD];
    const int row = blockIdx.x / HH;     // b*T + t
    const int h   = blockIdx.x % HH;
    const int t   = row % TT;
    const int d   = threadIdx.x;
    const int c   = h * DD + d;

    float y = cb[c];
    #pragma unroll
    for (int j = 0; j < CONVW; ++j) {
        int tj = t - (CONVW - 1) + j;
        if (tj >= 0)
            y += cw[c * CONVW + j] * pre[(size_t)(row - t + tj) * NBIG + colbase + c];
    }
    y = siluf_(y);
    sv[d] = y;
    __syncthreads();

    // RoPE
    const int i = (d < 32) ? d : d - 32;
    const int j = (i < 16) ? 2 * i : 2 * i - 32;     // inv_freq index
    const float invf = __expf(-(float)j * (9.21034037198f / 32.0f)); // ln(10000)/32
    const float ang = (float)t * invf;
    const float cs = cosf(ang), sn = sinf(ang);
    const float x1 = sv[2 * i], x2 = sv[2 * i + 1];
    float r = (d < 32) ? (x1 * cs - x2 * sn) : (x1 * sn + x2 * cs);

    // L2 norm over head dim
    red[d] = r * r;
    __syncthreads();
    float ss = 0.f;
    #pragma unroll
    for (int u = 0; u < DD; ++u) ss += red[u];
    r *= rsqrtf(ss + 1e-12f);

    out[(size_t)row * KD + c] = r;
}

// ---------------- depthwise conv(4)+SiLU for v ----------------
__global__ void conv_silu_v(const float* __restrict__ pre, const float* __restrict__ cw,
                            const float* __restrict__ cb, float* __restrict__ out)
{
    const int idx = blockIdx.x * 256 + threadIdx.x;     // over M*KD
    const int row = idx / KD;
    const int c   = idx % KD;
    const int t   = row % TT;
    float y = cb[c];
    #pragma unroll
    for (int j = 0; j < CONVW; ++j) {
        int tj = t - (CONVW - 1) + j;
        if (tj >= 0)
            y += cw[c * CONVW + j] * pre[(size_t)(row - t + tj) * NBIG + 2048 + c];
    }
    out[(size_t)row * KD + c] = siluf_(y);
}

// ---------------- beta / alpha heads ----------------
// grid = M blocks, block = 32 (16 beta heads + 16 alpha heads)
__global__ void beta_alpha(const float* __restrict__ x,
                           const float* __restrict__ Wb,  const float* __restrict__ bb,
                           const float* __restrict__ Wgk, const float* __restrict__ bgk,
                           const float* __restrict__ A_log, const float* __restrict__ dt_bias,
                           float* __restrict__ beta, float* __restrict__ alpha)
{
    const int row = blockIdx.x;
    const int tid = threadIdx.x;
    const int h   = tid & 15;
    const bool isA = tid >= 16;
    const float* W = isA ? (Wgk + (size_t)h * HID) : (Wb + (size_t)h * HID);
    const float* xr = x + (size_t)row * HID;
    float acc = 0.f;
    for (int k = 0; k < HID; ++k) acc = fmaf(xr[k], W[k], acc);
    if (!isA) {
        beta[row * HH + h] = sigmoidf_(acc + bb[h]);
    } else {
        float z = acc + bgk[h] + dt_bias[h];
        float sp = (z > 20.f) ? z : log1pf(__expf(z));
        alpha[row * HH + h] = __expf(-__expf(A_log[h]) * sp);
    }
}

// ---------------- gated delta-rule recurrent scan ----------------
// grid = B*H blocks, block = 64 (v column owner); S[:,v] lives in VGPRs
__global__ void delta_scan(const float* __restrict__ q, const float* __restrict__ k,
                           const float* __restrict__ v, const float* __restrict__ alpha,
                           const float* __restrict__ beta, const float* __restrict__ Dp,
                           float* __restrict__ o)
{
    __shared__ float qs[DD], ks[DD], vs[DD];
    const int bh = blockIdx.x;
    const int b  = bh / HH;
    const int h  = bh % HH;
    const int tid = threadIdx.x;
    const float dp = Dp[h];

    float S[DD];
    #pragma unroll
    for (int d = 0; d < DD; ++d) S[d] = 0.f;

    for (int t = 0; t < TT; ++t) {
        const int row = b * TT + t;
        const size_t base = (size_t)row * KD + h * DD;
        qs[tid] = q[base + tid];
        ks[tid] = k[base + tid];
        vs[tid] = v[base + tid];
        __syncthreads();

        const float a   = alpha[row * HH + h];
        const float bta = beta[row * HH + h];

        float ov = 0.f, kS = 0.f, qk = 0.f;
        #pragma unroll
        for (int d = 0; d < DD; ++d) {
            ov = fmaf(qs[d], S[d], ov);
            kS = fmaf(ks[d], S[d], kS);
            qk = fmaf(qs[d], ks[d], qk);
        }
        const float vt = vs[tid];
        #pragma unroll
        for (int d = 0; d < DD; ++d) {
            const float bk = bta * ks[d];
            S[d] = a * (S[d] - bk * kS) + bk * vt;
        }
        o[base + tid] = ov + dp * qk * vt;
        __syncthreads();
    }
}

// ---------------- RMSNorm + SiLU gate -> bf16 ----------------
// grid = B*T*H, block = 64
__global__ void norm_gate(const float* __restrict__ o, const float* __restrict__ pre,
                          const float* __restrict__ onorm_w,
                          unsigned short* __restrict__ og)
{
    __shared__ float red[DD];
    const int row = blockIdx.x / HH;
    const int h   = blockIdx.x % HH;
    const int d   = threadIdx.x;
    const size_t idx = (size_t)row * KD + h * DD + d;

    const float val = o[idx];
    red[d] = val * val;
    __syncthreads();
    float ss = 0.f;
    #pragma unroll
    for (int u = 0; u < DD; ++u) ss += red[u];
    const float on = val * rsqrtf(ss * (1.0f / DD) + EPSR) * onorm_w[d];
    const float g  = pre[(size_t)row * NBIG + 3072 + h * DD + d];
    og[idx] = f2bf(g * siluf_(on));
}

// ---------------- host side ----------------
extern "C" void kernel_launch(void* const* d_in, const int* in_sizes, int n_in,
                              void* d_out, int out_size, void* d_ws, size_t ws_size,
                              hipStream_t stream) {
    const float* x    = (const float*)d_in[0];
    const float* Wq   = (const float*)d_in[1];
    const float* Wk   = (const float*)d_in[2];
    const float* Wv   = (const float*)d_in[3];
    const float* Wg   = (const float*)d_in[4];
    const float* Wo   = (const float*)d_in[5];
    const float* Wb   = (const float*)d_in[6];
    const float* bb   = (const float*)d_in[7];
    const float* Wgk  = (const float*)d_in[8];
    const float* bgk  = (const float*)d_in[9];
    const float* cqw  = (const float*)d_in[10];
    const float* cqb  = (const float*)d_in[11];
    const float* ckw  = (const float*)d_in[12];
    const float* ckb  = (const float*)d_in[13];
    const float* cvw  = (const float*)d_in[14];
    const float* cvb  = (const float*)d_in[15];
    const float* A_log   = (const float*)d_in[16];
    const float* Dp      = (const float*)d_in[17];
    const float* dt_bias = (const float*)d_in[18];
    const float* onorm_w = (const float*)d_in[19];
    float* out = (float*)d_out;

    // workspace layout
    char* ws = (char*)d_ws;
    size_t off = 0;
    auto take = [&](size_t bytes) { char* p = ws + off; off = (off + bytes + 255) & ~(size_t)255; return p; };

    unsigned short* xbf    = (unsigned short*)take((size_t)MM * HID * 2);
    unsigned short* wqkvg  = (unsigned short*)take((size_t)NBIG * HID * 2);   // Wq|Wk|Wv|Wg rows
    unsigned short* wo_bf  = (unsigned short*)take((size_t)HID * KD * 2);
    float*          pre    = (float*)take((size_t)MM * NBIG * 4);             // q|k|v|g pre-activations
    float*          qb     = (float*)take((size_t)MM * KD * 4);
    float*          kb     = (float*)take((size_t)MM * KD * 4);
    float*          vb     = (float*)take((size_t)MM * KD * 4);
    float*          betab  = (float*)take((size_t)MM * HH * 4);
    float*          alphab = (float*)take((size_t)MM * HH * 4);
    float*          ob     = (float*)take((size_t)MM * KD * 4);
    unsigned short* og     = (unsigned short*)take((size_t)MM * KD * 2);
    (void)ws_size; (void)in_sizes; (void)n_in; (void)out_size;

    const int WN = HID * KD;   // 1M elements per weight matrix

    // 1) converts
    cvt_f32_bf16<<<(MM * HID + 255) / 256, 256, 0, stream>>>(x, xbf, MM * HID);
    cvt_f32_bf16<<<(WN + 255) / 256, 256, 0, stream>>>(Wq, wqkvg + 0 * (size_t)WN, WN);
    cvt_f32_bf16<<<(WN + 255) / 256, 256, 0, stream>>>(Wk, wqkvg + 1 * (size_t)WN, WN);
    cvt_f32_bf16<<<(WN + 255) / 256, 256, 0, stream>>>(Wv, wqkvg + 2 * (size_t)WN, WN);
    cvt_f32_bf16<<<(WN + 255) / 256, 256, 0, stream>>>(Wg, wqkvg + 3 * (size_t)WN, WN);
    cvt_f32_bf16<<<(WN + 255) / 256, 256, 0, stream>>>(Wo, wo_bf, WN);

    // 2) fused q|k|v|g projection GEMM (WMMA)
    wmma_gemm_bf16<<<dim3(NBIG / TILE_N, MM / TILE_M), 256, 0, stream>>>(
        xbf, wqkvg, pre, MM, NBIG, HID);

    // 3) beta / alpha heads (f32 VALU, tiny)
    beta_alpha<<<MM, 32, 0, stream>>>(x, Wb, bb, Wgk, bgk, A_log, dt_bias, betab, alphab);

    // 4) conv + SiLU (+RoPE +L2 for q/k)
    conv_rope_l2<<<MM * HH, DD, 0, stream>>>(pre, 0,    cqw, cqb, qb);
    conv_rope_l2<<<MM * HH, DD, 0, stream>>>(pre, 1024, ckw, ckb, kb);
    conv_silu_v <<<(MM * KD) / 256, 256, 0, stream>>>(pre, cvw, cvb, vb);

    // 5) gated delta-rule scan (f32, state in VGPRs)
    delta_scan<<<BB * HH, DD, 0, stream>>>(qb, kb, vb, alphab, betab, Dp, ob);

    // 6) RMSNorm + SiLU gate -> bf16 activations for output projection
    norm_gate<<<MM * HH, DD, 0, stream>>>(ob, pre, onorm_w, og);

    // 7) output projection GEMM (WMMA) -> d_out (f32)
    wmma_gemm_bf16<<<dim3(HID / TILE_N, MM / TILE_M), 256, 0, stream>>>(
        og, wo_bf, out, MM, HID, HID);
}
